// Layer_LP_6227702579357
// MI455X (gfx1250) — matrix operand order repaired and checked by reading, not compiled, not run on hardware
//
#include <hip/hip_runtime.h>
#include <hip/hip_bf16.h>

// Problem constants (from reference): B=16, T=8, Cin=Cout=128, H=W=32, K=3
#define BB    16
#define TT    8
#define CIN   128
#define COUT  128
#define HH_   32
#define WW_   32
#define KTOT  (CIN * 9)          // 1152  (k = tap*CIN + cin, tap = kh*3+kw)
#define NPAIR (KTOT / 2)         // 576 k-pairs per cout
#define HW    (HH_ * WW_)        // 1024

// Packed bf16 input: xq[bt][34 rows][36 cols][64 cin-pairs] as u32
// u32 = { bf16(cin=2p) , bf16(cin=2p+1)<<16 } at one spatial position.
#define XH    34
#define XW    36
#define XP    64                          // cin pairs
#define XROW  (XW * XP)                   // 2304 u32 per padded row
#define XBT   (XH * XROW)                 // 78336 u32 per bt  (313344 B)
#define XTOT  (BB * TT * XBT)             // 10,027,008 u32 (~40.1 MB)

typedef __attribute__((ext_vector_type(16))) __bf16 v16bf;
typedef __attribute__((ext_vector_type(8)))  float  v8f;

__device__ __forceinline__ unsigned short f2bf_rne(float f) {
    unsigned int u = __float_as_uint(f);
    unsigned int r = u + 0x7FFFu + ((u >> 16) & 1u);
    return (unsigned short)(r >> 16);
}

// ---------------------------------------------------------------------------
// Kernel 1: fold BN into weights/bias; weights -> bf16 pairs, k = tap*128+cin,
// laid out per-cout-contiguous:  Wq[cout * 576 + kp] = {k=2kp, k=2kp+1}
// ---------------------------------------------------------------------------
__global__ void snn_prepack_w(const float* __restrict__ w,
                              const float* __restrict__ bias,
                              const float* __restrict__ gamma,
                              const float* __restrict__ beta,
                              const float* __restrict__ mean,
                              const float* __restrict__ var,
                              unsigned int* __restrict__ Wq,
                              float* __restrict__ biasF) {
    int idx = blockIdx.x * blockDim.x + threadIdx.x;
    if (idx < COUT) {
        float inv = gamma[idx] * rsqrtf(var[idx] + 1e-5f);
        biasF[idx] = bias[idx] * inv + beta[idx] - mean[idx] * inv;
    }
    if (idx < NPAIR * COUT) {
        int cout = idx / NPAIR;
        int kp   = idx - cout * NPAIR;    // 0..575
        int tap  = kp >> 6;               // kh*3+kw
        int cp   = kp & 63;               // cin pair
        float inv = gamma[cout] * rsqrtf(var[cout] + 1e-5f);
        // reference layout: w[cout][cin][kh][kw] -> cout*1152 + cin*9 + tap
        float w0 = __builtin_nontemporal_load(&w[cout * KTOT + (2 * cp + 0) * 9 + tap]) * inv;
        float w1 = __builtin_nontemporal_load(&w[cout * KTOT + (2 * cp + 1) * 9 + tap]) * inv;
        Wq[idx] = (unsigned int)f2bf_rne(w0) |
                  ((unsigned int)f2bf_rne(w1) << 16);
    }
}

// ---------------------------------------------------------------------------
// Kernel 2: x (f32 NCHW) -> zero-padded, cin-pair-packed bf16 [bt][34][36][64].
// Writes every cell (incl. pads) each call -> deterministic. Writes coalesced.
// ---------------------------------------------------------------------------
__global__ void snn_prepack_x(const float* __restrict__ x,
                              unsigned int* __restrict__ xq) {
    int idx = blockIdx.x * blockDim.x + threadIdx.x;
    if (idx >= XTOT) return;
    int p  = idx & 63;                    // cin pair
    int r  = idx >> 6;
    int wp = r % XW;                      // padded col 0..35
    int r2 = r / XW;
    int hp = r2 % XH;                     // padded row 0..33
    int bt = r2 / XH;
    unsigned int v = 0;
    if (hp >= 1 && hp <= HH_ && wp >= 1 && wp <= WW_) {
        size_t base = ((size_t)bt * CIN + 2 * p) * HW + (hp - 1) * WW_ + (wp - 1);
        float f0 = __builtin_nontemporal_load(&x[base]);
        float f1 = __builtin_nontemporal_load(&x[base + HW]);
        v = (unsigned int)f2bf_rne(f0) | ((unsigned int)f2bf_rne(f1) << 16);
    }
    xq[idx] = v;
}

// ---------------------------------------------------------------------------
// Kernel 3: fused implicit-GEMM conv (bf16 WMMA) + bias + LIF scan + spikes.
// One wave => 16 couts x 16 pixels, all 8 timesteps live in registers.
// Per k-chunk: 2 b128 A-loads + 16 b128 B-loads issued as one clause, then
// 8 WMMAs -> partial loadcnt waits, loads of the next chunk overlap WMMAs.
// ---------------------------------------------------------------------------
__global__ void __launch_bounds__(256)
snn_conv_lif(const unsigned int* __restrict__ xq,
             const unsigned int* __restrict__ Wq,
             const float* __restrict__ biasF,
             const float* __restrict__ threshP,
             float* __restrict__ out) {
    const int lane     = threadIdx.x & 31;
    const int wave     = threadIdx.x >> 5;
    const int coutbase = wave * 16;
    const int lh       = (lane >= 16) ? 1 : 0;   // half-wave
    const int l15      = lane & 15;

    const int tile = blockIdx.x;                 // 0..63
    const int h    = tile >> 1;                  // output row 0..31
    const int w0   = (tile & 1) << 4;            // 0 or 16
    const int b    = blockIdx.y;                 // batch 0..15
    const int wpix = w0 + l15;                   // this lane's output column

    const float thresh = threshP[0];

    v8f acc[TT];
    #pragma unroll
    for (int t = 0; t < TT; ++t)
        acc[t] = (v8f){0.f,0.f,0.f,0.f,0.f,0.f,0.f,0.f};

    // A base: lane's cout block + half-wave pair offset (+8 pairs)
    const unsigned int* ap = Wq + (size_t)(coutbase + l15) * NPAIR + lh * 8;
    // B base: batch + half-wave pair offset (+8 pairs)
    const unsigned int* bp0 = xq + (size_t)(b * TT) * XBT + lh * 8;

    for (int tap = 0; tap < 9; ++tap) {          // (kh,kw) uniform
        const int kh = tap / 3;
        const int kw = tap - kh * 3;
        // padded coords -> always in-bounds, pad rows/cols are zeros
        const unsigned int* bpt = bp0 + (h + kh) * XROW + (wpix + kw) * XP;
        const unsigned int* apt = ap + tap * 64;

        for (int sub = 0; sub < 4; ++sub) {      // 32-cin k-chunk
            // ---- A fragment: VGPR j = K pair {0..7} (+lh*8) --------
            union { v16bf v; uint4 q[2]; } af;
            const unsigned int* a2 = apt + sub * 16;
            af.q[0] = *(const uint4*)(a2);
            af.q[1] = *(const uint4*)(a2 + 4);

            // ---- all 8 timestep B fragments first (one big load clause) ---
            const unsigned int* b2 = bpt + sub * 16;
            union { v16bf v; uint4 q[2]; } bf[TT];
            #pragma unroll
            for (int t = 0; t < TT; ++t) {       // imm offsets: t*313344(+16)
                bf[t].q[0] = *(const uint4*)(b2 + t * XBT);
                bf[t].q[1] = *(const uint4*)(b2 + t * XBT + 4);
            }
            // ---- then the 8 WMMAs: partial waits, pipelined with loads ----
            #pragma unroll
            for (int t = 0; t < TT; ++t)
                acc[t] = __builtin_amdgcn_wmma_f32_16x16x32_bf16(
                             false, af.v, false, bf[t].v,
                             (short)0, acc[t], false, false);
        }
    }

    // ---- epilogue: bias + LIF scan (mem*0.5 + y, hard reset) + store ------
    // D layout: VGPR r -> M(cout) = coutbase + r + (lh?8:0); N(pixel) = l15.
    #pragma unroll
    for (int r = 0; r < 8; ++r) {
        const int coutr = coutbase + r + (lh ? 8 : 0);
        const float br  = biasF[coutr];
        float mem = 0.0f;
        #pragma unroll
        for (int t = 0; t < TT; ++t) {
            float y = acc[t][r] + br;
            mem = mem * 0.5f + y;                       // TAU = 0.5
            float spike = (mem >= thresh) ? 1.0f : 0.0f;
            // write-once stream -> non-temporal, keep L2 for xq/Wq
            __builtin_nontemporal_store(
                spike,
                &out[(((size_t)(b * TT + t) * COUT + coutr) << 10)
                     + (h << 5) + wpix]);
            mem = (1.0f - spike) * mem;                 // hard reset
        }
    }
}

// ---------------------------------------------------------------------------
extern "C" void kernel_launch(void* const* d_in, const int* in_sizes, int n_in,
                              void* d_out, int out_size, void* d_ws, size_t ws_size,
                              hipStream_t stream) {
    const float* x      = (const float*)d_in[0];
    const float* w      = (const float*)d_in[1];
    const float* bias   = (const float*)d_in[2];
    const float* gamma  = (const float*)d_in[3];
    const float* beta   = (const float*)d_in[4];
    const float* mean   = (const float*)d_in[5];
    const float* var    = (const float*)d_in[6];
    const float* thresh = (const float*)d_in[7];
    float* out = (float*)d_out;

    // ws layout: Wq (288 KiB) | biasF (512 B) | xq (~40.1 MB), 16B aligned
    unsigned int* Wq    = (unsigned int*)d_ws;
    float*        biasF = (float*)((char*)d_ws + (size_t)NPAIR * COUT * 4);
    unsigned int* xq    = (unsigned int*)((char*)d_ws
                            + (size_t)NPAIR * COUT * 4 + COUT * 4);

    int npack = NPAIR * COUT;
    snn_prepack_w<<<(npack + 255) / 256, 256, 0, stream>>>(
        w, bias, gamma, beta, mean, var, Wq, biasF);

    snn_prepack_x<<<(XTOT + 255) / 256, 256, 0, stream>>>(x, xq);

    dim3 grid(HH_ * (WW_ / 16), BB);   // (64 pixel tiles, 16 batches)
    snn_conv_lif<<<grid, 256, 0, stream>>>(xq, Wq, biasF, thresh, out);
}
